// Seq2Seq_47193100648567
// MI455X (gfx1250) — matrix operand order repaired and checked by reading, not compiled, not run on hardware
//
#include <hip/hip_runtime.h>
#include <math.h>

// ---------------- CDNA5 WMMA types ----------------
typedef __attribute__((ext_vector_type(16))) _Float16 v16h;
typedef __attribute__((ext_vector_type(8)))  _Float16 v8h;
typedef __attribute__((ext_vector_type(8)))  float    v8f;

#define NV    50000
#define NE    512
#define NH    512
#define NG3   1536
#define NLIN  400
#define NTOUT 100
#define NB    64
#define NLMAX 400
#define NVT   3125   // NV / 16 output-column tiles

__device__ __forceinline__ v8f wmma_acc(v16h a, v16h b, v8f c) {
  // v_wmma_f32_16x16x32_f16 : D = A(16x32 f16) * B(32x16 f16) + C(16x16 f32)
  return __builtin_amdgcn_wmma_f32_16x16x32_f16(false, a, false, b, (short)0, c,
                                                false, false);
}

// A fragment (16x32 f16). base points at tile row 0; source is [M][K] row-major.
__device__ __forceinline__ v16h ld_a16(const _Float16* __restrict__ base, int ld,
                                       int k0, int lane) {
  int m = lane & 15, hi = (lane >> 4) & 1;
  const _Float16* p = base + (size_t)m * ld + k0 + hi * 8;
  v8h lo = *(const v8h*)(p);
  v8h hh = *(const v8h*)(p + 16);
  return __builtin_shufflevector(lo, hh, 0, 1, 2, 3, 4, 5, 6, 7,
                                 8, 9, 10, 11, 12, 13, 14, 15);
}

// A fragment sourced from fp32 with per-row scale (ctx = enc0 * a0 fusion).
__device__ __forceinline__ v16h ld_a32s(const float* __restrict__ base, int ld,
                                        int k0, float scale, int lane) {
  int m = lane & 15, hi = (lane >> 4) & 1;
  const float* p = base + (size_t)m * ld + k0 + hi * 8;
  v16h f;
#pragma unroll
  for (int i = 0; i < 8; ++i) {
    f[i]     = (_Float16)(p[i] * scale);
    f[i + 8] = (_Float16)(p[i + 16] * scale);
  }
  return f;
}

// B fragment (32x16 f16). Weights stored [N][K] row-major (i.e. W, used as W^T).
__device__ __forceinline__ v16h ld_b16(const _Float16* __restrict__ base, int ld,
                                       int k0, int lane) {
  int n = lane & 15, hi = (lane >> 4) & 1;
  const _Float16* p = base + (size_t)n * ld + k0 + hi * 8;
  v8h lo = *(const v8h*)(p);
  v8h hh = *(const v8h*)(p + 16);
  return __builtin_shufflevector(lo, hh, 0, 1, 2, 3, 4, 5, 6, 7,
                                 8, 9, 10, 11, 12, 13, 14, 15);
}

// ---------------- utility kernels ----------------
__global__ void k_cvt_f16(const float* __restrict__ s, _Float16* __restrict__ d, int n) {
  int i = blockIdx.x * blockDim.x + threadIdx.x;
  int st = gridDim.x * blockDim.x;
  for (; i < n; i += st) d[i] = (_Float16)s[i];
}

__global__ void k_zero_f32(float* __restrict__ p, int n) {
  int i = blockIdx.x * blockDim.x + threadIdx.x;
  int st = gridDim.x * blockDim.x;
  for (; i < n; i += st) p[i] = 0.0f;
}

__global__ void k_zero_f16(_Float16* __restrict__ p, int n) {
  int i = blockIdx.x * blockDim.x + threadIdx.x;
  int st = gridDim.x * blockDim.x;
  for (; i < n; i += st) p[i] = (_Float16)0.0f;
}

// encoder embedding gather: out[l][b][e] = (f16) emb[tok[l*B+b]][e]
__global__ void k_gather_enc(const float* __restrict__ emb, const int* __restrict__ toks,
                             _Float16* __restrict__ out, int total) {
  int i = blockIdx.x * blockDim.x + threadIdx.x;
  int st = gridDim.x * blockDim.x;
  for (; i < total; i += st) {
    int e = i & (NE - 1);
    int lb = i >> 9;                       // NE == 512
    int tok = toks[lb];
    out[i] = (_Float16)emb[(size_t)tok * NE + e];
  }
}

// decoder inputs: token = SOS(=1) at t==0 else target[t-1][b]
__global__ void k_gather_dec(const float* __restrict__ emb, const int* __restrict__ tgt,
                             _Float16* __restrict__ out, int total) {
  int i = blockIdx.x * blockDim.x + threadIdx.x;
  int st = gridDim.x * blockDim.x;
  for (; i < total; i += st) {
    int e = i & (NE - 1);
    int lb = i >> 9;
    int t = lb / NB, b = lb % NB;
    int tok = (t == 0) ? 1 : tgt[(t - 1) * NB + b];
    out[i] = (_Float16)emb[(size_t)tok * NE + e];
  }
}

// ---------------- encoder gi precompute (off the critical path) --------------
// gi[t][b][:] = enc_x[t][b] @ enc_Wih^T + bih for ALL 400 steps as one GEMM
// [25600 x 512] x [512 x 1536]. One wave = one M-tile x six N-tiles, so every
// A and B element is loaded exactly once. grid (NG3/96 = 16, 25600/16 = 1600).
__global__ void k_gi_all(const _Float16* __restrict__ x16,  // [NLIN*NB][NE]
                         const _Float16* __restrict__ Wih,  // [NG3][NE]
                         const float* __restrict__ bih,
                         _Float16* __restrict__ gi) {       // [NLIN*NB][NG3]
  int lane = threadIdx.x & 31;
  int n0 = blockIdx.x * 96;
  int m0 = blockIdx.y * 16;
  v8f acc[6] = {{}, {}, {}, {}, {}, {}};
  const _Float16* xa = x16 + (size_t)m0 * NE;
  for (int k0 = 0; k0 < NE; k0 += 32) {
    v16h a = ld_a16(xa, NE, k0, lane);
#pragma unroll
    for (int g = 0; g < 6; ++g)
      acc[g] = wmma_acc(a, ld_b16(Wih + (size_t)(n0 + g * 16) * NE, NE, k0, lane),
                        acc[g]);
  }
  int col = lane & 15, hi = (lane >> 4) & 1;
#pragma unroll
  for (int g = 0; g < 6; ++g) {
    int n = n0 + g * 16 + col;
    float bias = bih[n];
#pragma unroll
    for (int r = 0; r < 8; ++r) {
      int m = m0 + hi * 8 + r;
      gi[(size_t)m * NG3 + n] = (_Float16)(acc[g][r] + bias);
    }
  }
}

// ---------------- GRU step with precomputed gi (encoder fast path) -----------
// Only the sequential half (h @ Whh^T) remains: 3 tiles instead of 6.
__global__ void k_gru_step_pg(const _Float16* __restrict__ gi,   // [NB][NG3] (bih folded)
                              const _Float16* __restrict__ h16,
                              const float*   __restrict__ h32,
                              const _Float16* __restrict__ Whh,
                              const float* __restrict__ bhh,
                              float* __restrict__ h32o, _Float16* __restrict__ h16o,
                              float* __restrict__ enc0o /* nullable */) {
  int lane = threadIdx.x & 31;
  int j0 = blockIdx.x * 16;
  int m0 = blockIdx.y * 16;
  v8f ahr = {}, ahz = {}, ahn = {};
  const _Float16* ha = h16 + (size_t)m0 * NH;
  const _Float16* w_hr = Whh + (size_t)(j0) * NH;
  const _Float16* w_hz = Whh + (size_t)(NH + j0) * NH;
  const _Float16* w_hn = Whh + (size_t)(2 * NH + j0) * NH;
  for (int k0 = 0; k0 < NH; k0 += 32) {
    v16h ah = ld_a16(ha, NH, k0, lane);
    ahr = wmma_acc(ah, ld_b16(w_hr, NH, k0, lane), ahr);
    ahz = wmma_acc(ah, ld_b16(w_hz, NH, k0, lane), ahz);
    ahn = wmma_acc(ah, ld_b16(w_hn, NH, k0, lane), ahn);
  }
  int col = lane & 15, hi = (lane >> 4) & 1;
  int n = j0 + col;
  float b_hr = bhh[n], b_hz = bhh[NH + n], b_hn = bhh[2 * NH + n];
#pragma unroll
  for (int r = 0; r < 8; ++r) {
    int m = m0 + hi * 8 + r;
    float gir = (float)gi[(size_t)m * NG3 + n];
    float giz = (float)gi[(size_t)m * NG3 + NH + n];
    float gin = (float)gi[(size_t)m * NG3 + 2 * NH + n];
    float rr = 1.0f / (1.0f + expf(-(gir + ahr[r] + b_hr)));
    float zz = 1.0f / (1.0f + expf(-(giz + ahz[r] + b_hz)));
    float nn = tanhf(gin + rr * (ahn[r] + b_hn));
    float hprev = h32[(size_t)m * NH + n];
    float hnew = (1.0f - zz) * nn + zz * hprev;
    h32o[(size_t)m * NH + n] = hnew;
    h16o[(size_t)m * NH + n] = (_Float16)hnew;
    if (enc0o) enc0o[(size_t)m * NH + n] = hnew;
  }
}

// ---------------- fully fused GRU step (decoder / fallback) ------------------
__global__ void k_gru_step(const _Float16* __restrict__ x16,  // [NB][512]
                           const _Float16* __restrict__ h16,
                           const float*   __restrict__ h32,
                           const _Float16* __restrict__ Wih,  // [3H][512] K-major
                           const _Float16* __restrict__ Whh,  // [3H][NH]
                           const float* __restrict__ bih, const float* __restrict__ bhh,
                           float* __restrict__ h32o, _Float16* __restrict__ h16o,
                           float* __restrict__ enc0o /* nullable */) {
  int lane = threadIdx.x & 31;
  int j0 = blockIdx.x * 16;
  int m0 = blockIdx.y * 16;
  v8f air = {}, aiz = {}, ain = {}, ahr = {}, ahz = {}, ahn = {};
  const _Float16* xa = x16 + (size_t)m0 * NE;
  const _Float16* ha = h16 + (size_t)m0 * NH;
  const _Float16* w_ir = Wih + (size_t)(j0) * NE;
  const _Float16* w_iz = Wih + (size_t)(NH + j0) * NE;
  const _Float16* w_in = Wih + (size_t)(2 * NH + j0) * NE;
  const _Float16* w_hr = Whh + (size_t)(j0) * NH;
  const _Float16* w_hz = Whh + (size_t)(NH + j0) * NH;
  const _Float16* w_hn = Whh + (size_t)(2 * NH + j0) * NH;
  for (int k0 = 0; k0 < NE; k0 += 32) {
    v16h ax = ld_a16(xa, NE, k0, lane);
    v16h ah = ld_a16(ha, NH, k0, lane);
    air = wmma_acc(ax, ld_b16(w_ir, NE, k0, lane), air);
    aiz = wmma_acc(ax, ld_b16(w_iz, NE, k0, lane), aiz);
    ain = wmma_acc(ax, ld_b16(w_in, NE, k0, lane), ain);
    ahr = wmma_acc(ah, ld_b16(w_hr, NH, k0, lane), ahr);
    ahz = wmma_acc(ah, ld_b16(w_hz, NH, k0, lane), ahz);
    ahn = wmma_acc(ah, ld_b16(w_hn, NH, k0, lane), ahn);
  }
  int col = lane & 15, hi = (lane >> 4) & 1;
  int n = j0 + col;
  float b_ir = bih[n], b_iz = bih[NH + n], b_in = bih[2 * NH + n];
  float b_hr = bhh[n], b_hz = bhh[NH + n], b_hn = bhh[2 * NH + n];
#pragma unroll
  for (int r = 0; r < 8; ++r) {
    int m = m0 + hi * 8 + r;
    float gr = (air[r] + b_ir) + (ahr[r] + b_hr);
    float gz = (aiz[r] + b_iz) + (ahz[r] + b_hz);
    float rr = 1.0f / (1.0f + expf(-gr));
    float zz = 1.0f / (1.0f + expf(-gz));
    float nn = tanhf((ain[r] + b_in) + rr * (ahn[r] + b_hn));
    float hprev = h32[(size_t)m * NH + n];
    float hnew = (1.0f - zz) * nn + zz * hprev;
    h32o[(size_t)m * NH + n] = hnew;
    h16o[(size_t)m * NH + n] = (_Float16)hnew;
    if (enc0o) enc0o[(size_t)m * NH + n] = hnew;
  }
}

// ---------------- attention scores: [x,h] @ attn_W^T + attn_b ----------------
__global__ void k_scores(const _Float16* __restrict__ x16, const _Float16* __restrict__ h16,
                         const _Float16* __restrict__ Wa,  // [Lmax][1024]
                         const float* __restrict__ ba, float* __restrict__ scores) {
  int lane = threadIdx.x & 31;
  int n0 = blockIdx.x * 16, m0 = blockIdx.y * 16;
  v8f acc = {};
  const _Float16* wb = Wa + (size_t)n0 * (2 * NH);
  for (int k0 = 0; k0 < 2 * NH; k0 += 32) {
    v16h a = (k0 < NH) ? ld_a16(x16 + (size_t)m0 * NE, NE, k0, lane)
                       : ld_a16(h16 + (size_t)m0 * NH, NH, k0 - NH, lane);
    acc = wmma_acc(a, ld_b16(wb, 2 * NH, k0, lane), acc);
  }
  int col = lane & 15, hi = (lane >> 4) & 1;
  float bias = ba[n0 + col];
#pragma unroll
  for (int r = 0; r < 8; ++r) {
    int m = m0 + hi * 8 + r;
    scores[(size_t)m * NLMAX + n0 + col] = acc[r] + bias;
  }
}

// softmax over the row; only the column-0 probability survives (ctx = enc0*a0).
__global__ void k_softmax0(const float* __restrict__ scores, float* __restrict__ a0) {
  int b = blockIdx.x, lane = threadIdx.x;
  const float* row = scores + (size_t)b * NLMAX;
  float mx = -1e30f;
  for (int i = lane; i < NLMAX; i += 32) mx = fmaxf(mx, row[i]);
  for (int off = 16; off; off >>= 1) mx = fmaxf(mx, __shfl_xor(mx, off, 32));
  float s = 0.0f;
  for (int i = lane; i < NLMAX; i += 32) s += expf(row[i] - mx);
  for (int off = 16; off; off >>= 1) s += __shfl_xor(s, off, 32);
  if (lane == 0) a0[b] = expf(row[0] - mx) / s;
}

// combine: c = relu([x, enc0*a0] @ comb_W^T + comb_b) -> f16
__global__ void k_comb(const _Float16* __restrict__ x16, const float* __restrict__ enc0,
                       const float* __restrict__ a0, const _Float16* __restrict__ Wc,
                       const float* __restrict__ bc, _Float16* __restrict__ c16) {
  int lane = threadIdx.x & 31;
  int n0 = blockIdx.x * 16, m0 = blockIdx.y * 16;
  v8f acc = {};
  float sc = a0[m0 + (lane & 15)];
  const _Float16* wb = Wc + (size_t)n0 * (2 * NH);
  for (int k0 = 0; k0 < 2 * NH; k0 += 32) {
    v16h a = (k0 < NH) ? ld_a16(x16 + (size_t)m0 * NE, NE, k0, lane)
                       : ld_a32s(enc0 + (size_t)m0 * NH, NH, k0 - NH, sc, lane);
    acc = wmma_acc(a, ld_b16(wb, 2 * NH, k0, lane), acc);
  }
  int col = lane & 15, hi = (lane >> 4) & 1;
  float bias = bc[n0 + col];
#pragma unroll
  for (int r = 0; r < 8; ++r) {
    int m = m0 + hi * 8 + r;
    float v = fmaxf(acc[r] + bias, 0.0f);
    c16[(size_t)m * NH + n0 + col] = (_Float16)v;
  }
}

// ---------------- output projection with fused online-softmax stats ----------
__global__ void k_out_stats(const _Float16* __restrict__ h16,
                            const _Float16* __restrict__ Wo,   // [NV][NH]
                            const float* __restrict__ bo,
                            const int* __restrict__ tgt, int t,
                            float* __restrict__ pmax,          // [NB][NVT]
                            float* __restrict__ psum,          // [NB][NVT]
                            float* __restrict__ tlogit) {      // [NB]
  int lane = threadIdx.x & 31;
  int nt = blockIdx.x;          // 0..NVT-1
  int n0 = nt * 16;
  v8f acc[4] = {{}, {}, {}, {}};
  const _Float16* wb = Wo + (size_t)n0 * NH;
  for (int k0 = 0; k0 < NH; k0 += 32) {
    v16h b = ld_b16(wb, NH, k0, lane);
#pragma unroll
    for (int mt = 0; mt < 4; ++mt)
      acc[mt] = wmma_acc(ld_a16(h16 + (size_t)mt * 16 * NH, NH, k0, lane), b, acc[mt]);
  }
  int col = lane & 15, hi = (lane >> 4) & 1;
  float bias = bo[n0 + col];
#pragma unroll
  for (int mt = 0; mt < 4; ++mt) {
#pragma unroll
    for (int r = 0; r < 8; ++r) {
      int brow = mt * 16 + hi * 8 + r;
      float v = acc[mt][r] + bias;
      int y = tgt[t * NB + brow];
      if (y == n0 + col) tlogit[brow] = v;
      float mx = v;
      for (int off = 1; off < 16; off <<= 1) mx = fmaxf(mx, __shfl_xor(mx, off, 32));
      float s = expf(v - mx);
      for (int off = 1; off < 16; off <<= 1) s += __shfl_xor(s, off, 32);
      if (col == 0) {
        pmax[(size_t)brow * NVT + nt] = mx;
        psum[(size_t)brow * NVT + nt] = s;
      }
    }
  }
}

// merge per-tile online-softmax partials.
__global__ void k_loss_merge(const float* __restrict__ pmax, const float* __restrict__ psum,
                             const float* __restrict__ tlogit,
                             float* __restrict__ losses, int t) {
  int b = blockIdx.x, lane = threadIdx.x;
  const float* pm = pmax + (size_t)b * NVT;
  const float* ps = psum + (size_t)b * NVT;
  float M = -1e30f;
  for (int i = lane; i < NVT; i += 32) M = fmaxf(M, pm[i]);
  for (int off = 16; off; off >>= 1) M = fmaxf(M, __shfl_xor(M, off, 32));
  float S = 0.0f;
  for (int i = lane; i < NVT; i += 32) S += ps[i] * expf(pm[i] - M);
  for (int off = 16; off; off >>= 1) S += __shfl_xor(S, off, 32);
  if (lane == 0)
    losses[t * NB + b] = -(tlogit[b] - M - logf(S)) * (1.0f / (float)NB);
}

__global__ void k_reduce(const float* __restrict__ losses, float* __restrict__ out) {
  __shared__ float sm[256];
  int tid = threadIdx.x;
  float s = 0.0f;
  for (int i = tid; i < NTOUT * NB; i += 256) s += losses[i];
  sm[tid] = s;
  __syncthreads();
  for (int st = 128; st; st >>= 1) {
    if (tid < st) sm[tid] += sm[tid + st];
    __syncthreads();
  }
  if (tid == 0) out[0] = sm[0];
}

// ---------------- host orchestration ----------------
static inline int nblk(int n) {
  int g = (n + 255) / 256;
  return g > 4096 ? 4096 : g;
}

extern "C" void kernel_launch(void* const* d_in, const int* in_sizes, int n_in,
                              void* d_out, int out_size, void* d_ws, size_t ws_size,
                              hipStream_t stream) {
  (void)in_sizes; (void)n_in; (void)out_size;
  const float* emb     = (const float*)d_in[0];
  const float* enc_Wih = (const float*)d_in[1];
  const float* enc_Whh = (const float*)d_in[2];
  const float* enc_bih = (const float*)d_in[3];
  const float* enc_bhh = (const float*)d_in[4];
  const float* attn_W  = (const float*)d_in[5];
  const float* attn_b  = (const float*)d_in[6];
  const float* comb_W  = (const float*)d_in[7];
  const float* comb_b  = (const float*)d_in[8];
  const float* dec_Wih = (const float*)d_in[9];
  const float* dec_Whh = (const float*)d_in[10];
  const float* dec_bih = (const float*)d_in[11];
  const float* dec_bhh = (const float*)d_in[12];
  const float* out_W   = (const float*)d_in[13];
  const float* out_b   = (const float*)d_in[14];
  const int*   itok    = (const int*)d_in[15];
  const int*   ttok    = (const int*)d_in[16];

  char* wsb = (char*)d_ws;
  size_t o = 0;
  auto al = [&](size_t bytes) -> void* {
    void* p = wsb + o;
    o = (o + bytes + 255) & ~(size_t)255;
    return p;
  };
  _Float16* w_eih = (_Float16*)al((size_t)NG3 * NE * 2);
  _Float16* w_ehh = (_Float16*)al((size_t)NG3 * NH * 2);
  _Float16* w_dih = (_Float16*)al((size_t)NG3 * NH * 2);
  _Float16* w_dhh = (_Float16*)al((size_t)NG3 * NH * 2);
  _Float16* w_attn = (_Float16*)al((size_t)NLMAX * 2 * NH * 2);
  _Float16* w_comb = (_Float16*)al((size_t)NH * 2 * NH * 2);
  _Float16* w_out  = (_Float16*)al((size_t)NV * NH * 2);
  _Float16* encx   = (_Float16*)al((size_t)NLIN * NB * NE * 2);
  _Float16* decx   = (_Float16*)al((size_t)NTOUT * NB * NE * 2);
  float*    h32a   = (float*)al((size_t)NB * NH * 4);
  float*    h32b   = (float*)al((size_t)NB * NH * 4);
  _Float16* h16a   = (_Float16*)al((size_t)NB * NH * 2);
  _Float16* h16b   = (_Float16*)al((size_t)NB * NH * 2);
  float*    enc0   = (float*)al((size_t)NB * NH * 4);
  float*    scores = (float*)al((size_t)NB * NLMAX * 4);
  float*    a0     = (float*)al((size_t)NB * 4);
  _Float16* c16    = (_Float16*)al((size_t)NB * NH * 2);
  float*    pmax   = (float*)al((size_t)NB * NVT * 4);
  float*    psum   = (float*)al((size_t)NB * NVT * 4);
  float*    tlogit = (float*)al((size_t)NB * 4);
  float*    losses = (float*)al((size_t)NTOUT * NB * 4);
  if (o > ws_size) return;  // workspace too small; refuse to corrupt memory

  // optional encoder gi precompute buffer (78.6 MB): use it if ws allows
  size_t o_base = o;
  _Float16* gi_all = (_Float16*)al((size_t)NLIN * NB * NG3 * 2);
  bool use_pg = (o <= ws_size);
  if (!use_pg) { o = o_base; gi_all = nullptr; }

  float* h32[2] = {h32a, h32b};
  _Float16* h16[2] = {h16a, h16b};

  // weight conversions to f16 (K-major layouts already match W^T usage)
  k_cvt_f16<<<nblk(NG3 * NE), 256, 0, stream>>>(enc_Wih, w_eih, NG3 * NE);
  k_cvt_f16<<<nblk(NG3 * NH), 256, 0, stream>>>(enc_Whh, w_ehh, NG3 * NH);
  k_cvt_f16<<<nblk(NG3 * NH), 256, 0, stream>>>(dec_Wih, w_dih, NG3 * NH);
  k_cvt_f16<<<nblk(NG3 * NH), 256, 0, stream>>>(dec_Whh, w_dhh, NG3 * NH);
  k_cvt_f16<<<nblk(NLMAX * 2 * NH), 256, 0, stream>>>(attn_W, w_attn, NLMAX * 2 * NH);
  k_cvt_f16<<<nblk(NH * 2 * NH), 256, 0, stream>>>(comb_W, w_comb, NH * 2 * NH);
  k_cvt_f16<<<nblk(NV * NH), 256, 0, stream>>>(out_W, w_out, NV * NH);

  // embedding gathers
  k_gather_enc<<<nblk(NLIN * NB * NE), 256, 0, stream>>>(emb, itok, encx, NLIN * NB * NE);
  k_gather_dec<<<nblk(NTOUT * NB * NE), 256, 0, stream>>>(emb, ttok, decx, NTOUT * NB * NE);

  // h0 = 0
  k_zero_f32<<<nblk(NB * NH), 256, 0, stream>>>(h32a, NB * NH);
  k_zero_f16<<<nblk(NB * NH), 256, 0, stream>>>(h16a, NB * NH);

  dim3 gGru(NH / 16, NB / 16);
  int cur = 0;

  // -------- encoder --------
  if (use_pg) {
    // hoist the non-sequential half of all 400 GRU steps into one big GEMM
    k_gi_all<<<dim3(NG3 / 96, NLIN * NB / 16), 32, 0, stream>>>(encx, w_eih,
                                                                enc_bih, gi_all);
    for (int t = 0; t < NLIN; ++t) {
      k_gru_step_pg<<<gGru, 32, 0, stream>>>(
          gi_all + (size_t)t * NB * NG3, h16[cur], h32[cur], w_ehh, enc_bhh,
          h32[1 - cur], h16[1 - cur], (t == 0) ? enc0 : nullptr);
      cur ^= 1;
    }
  } else {
    for (int t = 0; t < NLIN; ++t) {
      k_gru_step<<<gGru, 32, 0, stream>>>(
          encx + (size_t)t * NB * NE, h16[cur], h32[cur], w_eih, w_ehh,
          enc_bih, enc_bhh, h32[1 - cur], h16[1 - cur],
          (t == 0) ? enc0 : nullptr);
      cur ^= 1;
    }
  }

  // -------- decoder: 100 teacher-forced steps --------
  dim3 gSc(NLMAX / 16, NB / 16);
  dim3 gCb(NH / 16, NB / 16);
  for (int t = 0; t < NTOUT; ++t) {
    const _Float16* xt = decx + (size_t)t * NB * NE;
    k_scores<<<gSc, 32, 0, stream>>>(xt, h16[cur], w_attn, attn_b, scores);
    k_softmax0<<<NB, 32, 0, stream>>>(scores, a0);
    k_comb<<<gCb, 32, 0, stream>>>(xt, enc0, a0, w_comb, comb_b, c16);
    k_gru_step<<<gGru, 32, 0, stream>>>(c16, h16[cur], h32[cur], w_dih, w_dhh,
                                        dec_bih, dec_bhh, h32[1 - cur], h16[1 - cur],
                                        nullptr);
    cur ^= 1;
    k_out_stats<<<NVT, 32, 0, stream>>>(h16[cur], w_out, out_b, ttok, t,
                                        pmax, psum, tlogit);
    k_loss_merge<<<NB, 32, 0, stream>>>(pmax, psum, tlogit, losses, t);
  }

  k_reduce<<<1, 256, 0, stream>>>(losses, (float*)d_out);
}
